// Knn_4887672783539
// MI455X (gfx1250) — matrix operand order repaired and checked by reading, not compiled, not run on hardware
//
#include <hip/hip_runtime.h>

typedef __attribute__((ext_vector_type(16))) __bf16 v16bf;
typedef __attribute__((ext_vector_type(8)))  float  v8f;

#define NQ   32768
#define NK   8192
#define DIM  64
#define TOPK 16
#define NT   (NQ / 16)      // 2048 query tiles
#define KPW  32             // keys per wave (2 B-tiles)

// ---------------- kernel 0a: query squared norms ----------------
__global__ void qnorm_kernel(const float* __restrict__ q,
                             float* __restrict__ q2, int n) {
    int m = blockIdx.x * blockDim.x + threadIdx.x;
    if (m >= n) return;
    const float4* row = (const float4*)(q + (size_t)m * DIM);
    float s = 0.f;
#pragma unroll
    for (int i = 0; i < DIM / 4; ++i) {
        float4 v = row[i];
        s = fmaf(v.x, v.x, s); s = fmaf(v.y, v.y, s);
        s = fmaf(v.z, v.z, s); s = fmaf(v.w, v.w, s);
    }
    q2[m] = s;
}

// ---------------- kernel 0b: pre-swizzle queries into bf16 A-fragment order --
// qswz[(t*32 + lane)*32 + e]: e=0..15 -> a0 elements, e=16..31 -> a1 elements.
// A (16x32 bf16) layout: lane l holds row l&15, K(e) = (e>>3)*16 + (l>=16?8:0) + (e&7)
__global__ void qswz_kernel(const float* __restrict__ q,
                            __bf16* __restrict__ qswz) {
    int tid = blockIdx.x * blockDim.x + threadIdx.x;   // t*32 + l
    if (tid >= NT * 32) return;
    int l = tid & 31;
    int t = tid >> 5;
    int off8 = (l >= 16) ? 8 : 0;
    const float* qrow = q + (size_t)(t * 16 + (l & 15)) * DIM;
    __bf16 tmp[32];
#pragma unroll
    for (int g = 0; g < 2; ++g)
#pragma unroll
        for (int w = 0; w < 8; ++w) {
            tmp[g * 8 + w]      = (__bf16)qrow[g * 16 + off8 + w];        // a0
            tmp[16 + g * 8 + w] = (__bf16)qrow[32 + g * 16 + off8 + w];   // a1
        }
    uint4* dst = (uint4*)(qswz + (size_t)tid * 32);
    const uint4* src = (const uint4*)tmp;
    dst[0] = src[0]; dst[1] = src[1]; dst[2] = src[2]; dst[3] = src[3];
}

// branch-guarded unrolled insert into ascending 16-entry register list
__device__ __forceinline__ void insert16(unsigned long long (&lst)[TOPK],
                                         unsigned long long key) {
#pragma unroll
    for (int i = TOPK - 1; i >= 1; --i) {
        unsigned long long lo = lst[i - 1];
        lst[i] = (key < lst[i]) ? ((key < lo) ? lo : key) : lst[i];
    }
    lst[0] = (key < lst[0]) ? key : lst[0];
}

// WGP-scope prefetch (scope field omitted -> SCOPE=0=WGP: fills all cache
// levels, unlike __builtin_prefetch which only reaches SE/SYS scope -> GL2).
__device__ __forceinline__ void prefetch_wgp(const void* p) {
    unsigned long long addr = (unsigned long long)p;
    asm volatile("global_prefetch_b8 %0, off" : : "v"(addr));
}

// ---------------- kernel 1: fused WMMA GEMM + per-key top-16 ----------------
// Wave owns 32 keys (two B tiles fixed in registers); sweeps all query tiles.
// C fragment column n lives in lanes n and n+16 -> per-lane register top-k.
__launch_bounds__(256)
__global__ void knn_topk_kernel(const float* __restrict__ keys,
                                const __bf16* __restrict__ qswz,
                                const float* __restrict__ q2,
                                int* __restrict__ out_idx,
                                float* __restrict__ out_idx_f) {
    __shared__ unsigned long long lmerge[8][32][TOPK];  // 32 KB, reused twice

    const int lane        = threadIdx.x & 31;
    const int waveInBlock = threadIdx.x >> 5;
    const int wave        = blockIdx.x * (blockDim.x >> 5) + waveInBlock;
    const int keyBase     = wave * KPW;
    const int nLocal      = lane & 15;
    const bool hi         = lane >= 16;
    const int off8        = hi ? 8 : 0;    // A-frag / C-frag row-half offset
    const int boff        = hi ? 16 : 0;   // unused; B-frag K-half offset is off8

    // ---- B fragments for two key tiles (bf16), plus ||key||^2 per tile
    const float* kr0 = keys + (size_t)(keyBase + nLocal) * DIM;
    const float* kr1 = kr0 + 16 * DIM;
    v16bf b00, b01, b10, b11;
    float k2h0 = 0.f, k2h1 = 0.f;
#pragma unroll
    for (int j = 0; j < 16; ++j) {
        float x0 = kr0[(j >> 3) * 16 + off8 + (j & 7)], y0 = kr0[32 + (j >> 3) * 16 + off8 + (j & 7)];
        float x1 = kr1[(j >> 3) * 16 + off8 + (j & 7)], y1 = kr1[32 + (j >> 3) * 16 + off8 + (j & 7)];
        b00[j] = (__bf16)x0; b01[j] = (__bf16)y0;
        b10[j] = (__bf16)x1; b11[j] = (__bf16)y1;
        k2h0 = fmaf(x0, x0, k2h0); k2h0 = fmaf(y0, y0, k2h0);
        k2h1 = fmaf(x1, x1, k2h1); k2h1 = fmaf(y1, y1, k2h1);
    }
    const float k2_0 = k2h0 + __shfl_xor(k2h0, 16, 32);
    const float k2_1 = k2h1 + __shfl_xor(k2h1, 16, 32);

    // ---- per-lane top-16 lists, ascending, packed (dist_bits<<32 | qidx)
    unsigned long long lst0[TOPK], lst1[TOPK];
#pragma unroll
    for (int j = 0; j < TOPK; ++j) { lst0[j] = ~0ull; lst1[j] = ~0ull; }
    unsigned thr0 = 0xFFFFFFFFu, thr1 = 0xFFFFFFFFu;  // dist bits of lst[15]

    for (int t = 0; t < NT; ++t) {
        // A fragments: 64 contiguous bytes per lane, pre-swizzled bf16
        const __bf16* ab = qswz + ((size_t)t * 32 + lane) * 32;
        v16bf a0 = *(const v16bf*)ab;
        v16bf a1 = *(const v16bf*)(ab + 16);

        // prefetch a future tile of the query stream into near caches
        int tp = (t + 8 < NT) ? (t + 8) : t;
        prefetch_wgp(qswz + ((size_t)tp * 32 + lane) * 32);

        // query norms for this lane's 8 candidate rows (broadcast loads)
        float q2v[8];
        *(float4*)(q2v)     = *(const float4*)(q2 + t * 16 + off8);
        *(float4*)(q2v + 4) = *(const float4*)(q2 + t * 16 + off8 + 4);
        float base0[8], base1[8];
#pragma unroll
        for (int r = 0; r < 8; ++r) {
            base0[r] = k2_0 + q2v[r];
            base1[r] = k2_1 + q2v[r];
        }

        // dot(query_m, key_n) in f32 for both key tiles
        v8f c0 = {}, c1 = {};
        c0 = __builtin_amdgcn_wmma_f32_16x16x32_bf16(false, a0, false, b00,
                                                     (short)0, c0, false, false);
        c1 = __builtin_amdgcn_wmma_f32_16x16x32_bf16(false, a0, false, b10,
                                                     (short)0, c1, false, false);
        c0 = __builtin_amdgcn_wmma_f32_16x16x32_bf16(false, a1, false, b01,
                                                     (short)0, c0, false, false);
        c1 = __builtin_amdgcn_wmma_f32_16x16x32_bf16(false, a1, false, b11,
                                                     (short)0, c1, false, false);

        // epilogue: d2 = ||k||^2 + ||q||^2 - 2*dot ; thresholded insert
#pragma unroll
        for (int r = 0; r < 8; ++r) {
            unsigned qidx = (unsigned)(t * 16 + off8 + r);
            float d0 = fmaxf(fmaf(-2.f, c0[r], base0[r]), 0.f);
            unsigned db0 = __float_as_uint(d0);
            if (db0 <= thr0) {
                insert16(lst0, ((unsigned long long)db0 << 32) | qidx);
                thr0 = (unsigned)(lst0[TOPK - 1] >> 32);
            }
            float d1 = fmaxf(fmaf(-2.f, c1[r], base1[r]), 0.f);
            unsigned db1 = __float_as_uint(d1);
            if (db1 <= thr1) {
                insert16(lst1, ((unsigned long long)db1 << 32) | qidx);
                thr1 = (unsigned)(lst1[TOPK - 1] >> 32);
            }
        }
    }

    // ---- merge lane n and lane n+16 (disjoint query halves), two phases
#pragma unroll
    for (int phase = 0; phase < 2; ++phase) {
#pragma unroll
        for (int j = 0; j < TOPK; ++j)
            lmerge[waveInBlock][lane][j] = phase ? lst1[j] : lst0[j];
        __syncthreads();
        if (!hi) {
            const unsigned long long* La = lmerge[waveInBlock][lane];
            const unsigned long long* Lb = lmerge[waveInBlock][lane + 16];
            int ia = 0, ib = 0;
            const int n = keyBase + phase * 16 + lane;
            for (int j = 0; j < TOPK; ++j) {
                unsigned long long va = La[ia], vb = Lb[ib];
                bool takeA = va < vb;
                unsigned long long v = takeA ? va : vb;
                if (takeA) ++ia; else ++ib;
                int qi = (int)(unsigned)(v & 0xffffffffull);
                out_idx[n * TOPK + j]   = qi;
                out_idx_f[n * TOPK + j] = (float)qi;
            }
        }
        __syncthreads();
    }
}

// ---------------- kernel 2: exact fp32 distance recompute ----------------
__global__ void exact_dist_kernel(const float* __restrict__ keys,
                                  const float* __restrict__ queries,
                                  const int* __restrict__ idx,
                                  float* __restrict__ out_d, int total) {
    int t = blockIdx.x * blockDim.x + threadIdx.x;
    if (t >= total) return;
    int n  = t / TOPK;
    int qi = idx[t];
    const float4* kr = (const float4*)(keys + (size_t)n * DIM);
    const float4* qr = (const float4*)(queries + (size_t)qi * DIM);
    float s = 0.f;
#pragma unroll
    for (int i = 0; i < DIM / 4; ++i) {
        float4 a = kr[i], b = qr[i];
        float dx = a.x - b.x, dy = a.y - b.y, dz = a.z - b.z, dw = a.w - b.w;
        s = fmaf(dx, dx, s); s = fmaf(dy, dy, s);
        s = fmaf(dz, dz, s); s = fmaf(dw, dw, s);
    }
    out_d[t] = s;
}

// ---------------- launcher ----------------
extern "C" void kernel_launch(void* const* d_in, const int* in_sizes, int n_in,
                              void* d_out, int out_size, void* d_ws, size_t ws_size,
                              hipStream_t stream) {
    const float* keys    = (const float*)d_in[0];   // [8192, 64]
    const float* queries = (const float*)d_in[1];   // [32768, 64]
    float* out = (float*)d_out;                     // [8192*16 idx | 8192*16 dist]

    // workspace layout: q2 (128KB) | ws_idx (512KB) | qswz (4MB)
    float*  q2     = (float*)d_ws;
    int*    ws_idx = (int*)((char*)d_ws + NQ * sizeof(float));
    __bf16* qswz   = (__bf16*)((char*)d_ws + NQ * sizeof(float)
                                           + NK * TOPK * sizeof(int));

    qnorm_kernel<<<(NQ + 255) / 256, 256, 0, stream>>>(queries, q2, NQ);
    qswz_kernel<<<(NT * 32 + 255) / 256, 256, 0, stream>>>(queries, qswz);

    const int waves  = NK / KPW;      // 256 waves, 32 keys each
    const int blocks = waves / 8;     // 256 threads = 8 waves per block
    knn_topk_kernel<<<blocks, 256, 0, stream>>>(keys, qswz, q2, ws_idx, out);

    const int total = NK * TOPK;      // 131072 (key, neighbor) pairs
    exact_dist_kernel<<<(total + 255) / 256, 256, 0, stream>>>(
        keys, queries, ws_idx, out + total, total);
}
